// SimpleAttentionNet_78580721647676
// MI455X (gfx1250) — matrix operand order repaired and checked by reading, not compiled, not run on hardware
//
#include <hip/hip_runtime.h>

typedef __attribute__((ext_vector_type(16))) _Float16 v16h;
typedef __attribute__((ext_vector_type(8)))  _Float16 v8h;
typedef __attribute__((ext_vector_type(8)))  float    v8f;
typedef __attribute__((ext_vector_type(4)))  unsigned int u32x4;
typedef __attribute__((ext_vector_type(8)))  unsigned int u32x8;

#define NB  8
#define SEQ 2048
#define FD  512
#define DD  128
#define NC  1000

static __device__ __forceinline__ v16h cat16(v8h lo, v8h hi) {
  return __builtin_shufflevector(lo, hi, 0,1,2,3,4,5,6,7,8,9,10,11,12,13,14,15);
}

static __device__ __forceinline__ v8f wmma16(v16h a, v16h b, v8f c) {
  // D = A(16x32 f16) * B(32x16 f16) + C(16x16 f32)
  return __builtin_amdgcn_wmma_f32_16x16x32_f16(false, a, false, b, (short)0, c, false, false);
}

// LDS byte address (as used by DS / the TDM D#.lds_addr) of a __shared__ object.
typedef __attribute__((address_space(3))) const char as3char;
static __device__ __forceinline__ unsigned lds_addr_of(const void* p) {
  return (unsigned)(size_t)(as3char*)p;
}

// Issue one 2D TDM tile load: D# group0 built from (lds_addr, global_addr),
// group1 passed in (dims/strides/pad preconfigured).
static __device__ __forceinline__ void tdm_load(unsigned lds, unsigned long long ga,
                                                u32x8 g1) {
  u32x4 g0;
  g0[0] = 1u;                                            // count=1 (valid D#)
  g0[1] = lds;                                           // lds_addr
  g0[2] = (unsigned)ga;                                  // global_addr lo
  g0[3] = ((unsigned)(ga >> 32) & 0x01FFFFFFu) | (2u << 30);  // addr hi | type=2
  asm volatile("tensor_load_to_lds %0, %1" :: "s"(g0), "s"(g1) : "memory");
}

// ---------------------------------------------------------------------------
// Prep kernel 1: x f32 -> f16, 8 elements per thread
// ---------------------------------------------------------------------------
__global__ __launch_bounds__(256) void cvt_x_kernel(
    const float* __restrict__ x, _Float16* __restrict__ xh)
{
  const size_t i = ((size_t)blockIdx.x * 256 + threadIdx.x) * 8;
  float4 a = *(const float4*)&x[i];
  float4 b = *(const float4*)&x[i + 4];
  v8h h;
  h[0]=(_Float16)a.x; h[1]=(_Float16)a.y; h[2]=(_Float16)a.z; h[3]=(_Float16)a.w;
  h[4]=(_Float16)b.x; h[5]=(_Float16)b.y; h[6]=(_Float16)b.z; h[7]=(_Float16)b.w;
  *(v8h*)&xh[i] = h;
}

// ---------------------------------------------------------------------------
// Prep kernel 2: W [F,D] f32 -> W^T [D,F] f16
// ---------------------------------------------------------------------------
__global__ __launch_bounds__(256) void cvt_wT_kernel(
    const float* __restrict__ W, _Float16* __restrict__ WT)
{
  const int i = blockIdx.x * 256 + threadIdx.x;   // 512*128 = 65536 threads
  const int n = i & (DD - 1), f = i >> 7;
  WT[(size_t)n * FD + f] = (_Float16)W[i];
}

// ---------------------------------------------------------------------------
// Kernel A: projection GEMM, LDS-free. One wave = 16(M) x 16(N) tile, K=512
// fully unrolled (16 WMMAs); A/B fragments are direct global b128 loads.
// transposed==0: out row-major [B*S, D] (q, k);  ==1: out as V^T [B, D, S].
// ---------------------------------------------------------------------------
__global__ __launch_bounds__(256) void qkv_gemm_kernel(
    const _Float16* __restrict__ xh, const _Float16* __restrict__ WT,
    const float* __restrict__ bias, _Float16* __restrict__ out, int transposed)
{
  const int tid  = threadIdx.x;
  const int lane = tid & 31, w = tid >> 5;
  const int hi   = lane >> 4, n16 = lane & 15;
  const int row0 = blockIdx.x * 16;
  const int ng   = w * 16 + n16;

  const _Float16* xr = xh + (size_t)(row0 + n16) * FD;   // A: row = lane%16
  const _Float16* wr = WT + (size_t)ng * FD;             // B: col = lane%16, K contiguous

  v8f c = {};
#pragma unroll
  for (int kc = 0; kc < 16; ++kc) {
    const int k0 = kc * 32;
    v16h a  = cat16(*(const v8h*)&xr[k0 + 8 * hi],
                    *(const v8h*)&xr[k0 + 16 + 8 * hi]);
    v16h bm = cat16(*(const v8h*)&wr[k0 + 8 * hi],
                    *(const v8h*)&wr[k0 + 16 + 8 * hi]);
    c = wmma16(a, bm, c);
  }

  const float bb = bias[ng];
  if (!transposed) {
#pragma unroll
    for (int e = 0; e < 8; ++e) {
      int m = 8 * hi + e;
      out[(size_t)(row0 + m) * DD + ng] = (_Float16)(c[e] + bb);
    }
  } else {
    const int bi = row0 >> 11;                 // batch
    const int s0 = (row0 & 2047) + 8 * hi;     // seq base (16B aligned)
    v8h pk;
#pragma unroll
    for (int e = 0; e < 8; ++e) pk[e] = (_Float16)(c[e] + bb);
    *(v8h*)&out[((size_t)bi * DD + ng) * SEQ + s0] = pk;
  }
}

// ---------------------------------------------------------------------------
// Kernel B: fused flash attention + mean over S, with DOUBLE-BUFFERED TDM
// staging. Wave 0 issues tensor_load_to_lds for chunk jc+1 into buffer
// (jc+1)&1, then s_wait_tensorcnt(2): TDM ops complete in order per wave, so
// <=2 outstanding proves chunk jc's two tiles have landed while chunk jc+1
// streams in behind the compute. The end-of-iteration barrier guarantees all
// waves finished reading buffer jc&1 before iteration jc+1 overwrites it.
// D# pad fields reproduce the padded LDS strides (K: 64DW rows + 4DW pad ->
// stride 136 halves; V^T: 16DW rows + 4DW pad -> stride 40 halves).
// ---------------------------------------------------------------------------
__global__ __launch_bounds__(256) void attn_kernel(
    const _Float16* __restrict__ q, const _Float16* __restrict__ kmat,
    const _Float16* __restrict__ vT, float* __restrict__ partial)
{
  __shared__ _Float16 sK[2][32 * 136];   // K rows: [key][d], stride 136
  __shared__ _Float16 sV[2][128 * 40];   // V^T rows: [d][key], stride 40
  __shared__ _Float16 sP[8 * 16 * 40];   // per-wave P tile (C->A layout bounce)
  __shared__ float    sRed[8 * 128];     // cross-wave mean reduction
  const int tid  = threadIdx.x;
  const int lane = tid & 31, w = tid >> 5;
  const int hi   = lane >> 4, n16 = lane & 15;
  const int b    = blockIdx.x >> 4;
  const int qb   = blockIdx.x & 15;
  const int qr0  = qb * 128 + w * 16;

  v16h aQ[4];                             // Q tile, 16 rows x 128 d, A-fragments
  {
    const _Float16* qrow = q + ((size_t)(b * SEQ + qr0 + n16)) * DD;
#pragma unroll
    for (int dt = 0; dt < 4; ++dt)
      aQ[dt] = cat16(*(const v8h*)&qrow[dt * 32 + 8 * hi],
                     *(const v8h*)&qrow[dt * 32 + 16 + 8 * hi]);
  }

  float mrun[8], lrun[8];
  v8f   acc[8] = {};
#pragma unroll
  for (int e = 0; e < 8; ++e) { mrun[e] = -1e30f; lrun[e] = 0.0f; }

  _Float16* myP = &sP[w * 16 * 40];

  // ---- loop-invariant TDM descriptor pieces -------------------------------
  const unsigned sK_lds = lds_addr_of(&sK[0][0]);
  const unsigned sV_lds = lds_addr_of(&sV[0][0]);
  const unsigned sK_bufsz = 32 * 136 * 2;    // bytes per K buffer
  const unsigned sV_bufsz = 128 * 40 * 2;    // bytes per V buffer
  // D# group1: [31:25]=pad_amount-1(4DW->3) [24:22]=pad_interval [20]=pad_en
  // [17:16]=data_size(2B->1); then dims/tile/strides in data_size units.
  const u32x8 g1k = { (1u<<16) | (1u<<20) | (5u<<22) | (3u<<25),  // pad every 64DW
                      128u << 16,        // tensor_dim0 = 128
                      32u  << 16,        // tensor_dim1 = 32
                      128u << 16,        // tile_dim0  = 128
                      32u,               // tile_dim1  = 32
                      128u,              // tensor_dim0_stride = 128
                      0u, 0u };
  const u32x8 g1v = { (1u<<16) | (1u<<20) | (3u<<22) | (3u<<25),  // pad every 16DW
                      32u  << 16,        // tensor_dim0 = 32
                      128u << 16,        // tensor_dim1 = 128
                      32u  << 16,        // tile_dim0  = 32
                      128u,              // tile_dim1  = 128
                      2048u,             // tensor_dim0_stride = 2048
                      0u, 0u };
  const unsigned long long kbase =
      (unsigned long long)(size_t)kmat + ((size_t)b * SEQ) * DD * 2;
  const unsigned long long vbase =
      (unsigned long long)(size_t)vT + ((size_t)b * DD) * SEQ * 2;

  // prologue: start chunk 0 into buffer 0
  if (w == 0) {
    tdm_load(sK_lds, kbase, g1k);
    tdm_load(sV_lds, vbase, g1v);
  }

  for (int jc = 0; jc < 64; ++jc) {
    if (w == 0) {                          // wave-uniform: only wave 0 drives TDM
      if (jc < 63) {                       // prefetch chunk jc+1 into other buffer
        const unsigned nb = (unsigned)(jc + 1) & 1u;
        const unsigned long long ka = kbase + (unsigned long long)(jc + 1) * 32 * DD * 2;
        const unsigned long long va = vbase + (unsigned long long)(jc + 1) * 32 * 2;
        tdm_load(sK_lds + nb * sK_bufsz, ka, g1k);
        tdm_load(sV_lds + nb * sV_bufsz, va, g1v);
        __builtin_amdgcn_s_wait_tensorcnt(2);   // chunk jc landed; jc+1 in flight
      } else {
        __builtin_amdgcn_s_wait_tensorcnt(0);   // last chunk landed
      }
    }
    __syncthreads();                       // LDS tiles for chunk jc visible to all
    const _Float16* cK = &sK[jc & 1][0];
    const _Float16* cV = &sV[jc & 1][0];

    // logits: 16 q-rows x 32 keys (two 16x16 tiles), reduce over d=128
    v8f cl0 = {}, cl1 = {};
#pragma unroll
    for (int dt = 0; dt < 4; ++dt) {
      v16h b0 = cat16(*(const v8h*)&cK[n16 * 136 + dt * 32 + 8 * hi],
                      *(const v8h*)&cK[n16 * 136 + dt * 32 + 16 + 8 * hi]);
      cl0 = wmma16(aQ[dt], b0, cl0);
      v16h b1 = cat16(*(const v8h*)&cK[(16 + n16) * 136 + dt * 32 + 8 * hi],
                      *(const v8h*)&cK[(16 + n16) * 136 + dt * 32 + 16 + 8 * hi]);
      cl1 = wmma16(aQ[dt], b1, cl1);
    }

    // online softmax per q-row (rows live as element e, lo/hi lane groups)
#pragma unroll
    for (int e = 0; e < 8; ++e) {
      float t0 = cl0[e], t1 = cl1[e];
      float tm = fmaxf(t0, t1);
#pragma unroll
      for (int off = 8; off >= 1; off >>= 1) tm = fmaxf(tm, __shfl_xor(tm, off, 32));
      float mnew  = fmaxf(mrun[e], tm);
      float scale = __expf(mrun[e] - mnew);
      float p0 = __expf(t0 - mnew), p1 = __expf(t1 - mnew);
      float rs = p0 + p1;
#pragma unroll
      for (int off = 8; off >= 1; off >>= 1) rs += __shfl_xor(rs, off, 32);
      lrun[e] = lrun[e] * scale + rs;
      mrun[e] = mnew;
#pragma unroll
      for (int nt = 0; nt < 8; ++nt) acc[nt][e] *= scale;
      myP[(8 * hi + e) * 40 + n16]      = (_Float16)p0;   // C-layout -> LDS
      myP[(8 * hi + e) * 40 + 16 + n16] = (_Float16)p1;
    }

    // reload P in A-layout, accumulate P(16x32) * V(32x128)
    v16h aP = cat16(*(const v8h*)&myP[n16 * 40 + 8 * hi],
                    *(const v8h*)&myP[n16 * 40 + 16 + 8 * hi]);
#pragma unroll
    for (int nt = 0; nt < 8; ++nt) {
      v16h bv = cat16(*(const v8h*)&cV[(nt * 16 + n16) * 40 + 8 * hi],
                      *(const v8h*)&cV[(nt * 16 + n16) * 40 + 16 + 8 * hi]);
      acc[nt] = wmma16(aP, bv, acc[nt]);
    }
    __syncthreads();    // all reads of buffer jc&1 done before jc+1 overwrites it
  }

  // normalize by softmax denominator, then column-sum the 16 q-rows
#pragma unroll
  for (int e = 0; e < 8; ++e) {
    float inv = 1.0f / lrun[e];
#pragma unroll
    for (int nt = 0; nt < 8; ++nt) acc[nt][e] *= inv;
  }
#pragma unroll
  for (int nt = 0; nt < 8; ++nt) {
    float cs = 0.0f;
#pragma unroll
    for (int e = 0; e < 8; ++e) cs += acc[nt][e];
    cs += __shfl_xor(cs, 16, 32);                 // merge rows 0-7 with 8-15
    if (lane < 16) sRed[w * 128 + nt * 16 + lane] = cs;
  }
  __syncthreads();
  if (tid < 128) {                                 // merge 8 waves deterministically
    float s = 0.0f;
#pragma unroll
    for (int w2 = 0; w2 < 8; ++w2) s += sRed[w2 * 128 + tid];
    partial[(size_t)blockIdx.x * 128 + tid] = s;
  }
}

// ---------------------------------------------------------------------------
// Kernel C: mean commutes with the final linear layer:
//   out[b] = (mean_s attn_out[b]) @ Wl + bl     -> only 8x128x1000
// ---------------------------------------------------------------------------
__global__ __launch_bounds__(256) void final_kernel(
    const float* __restrict__ partial, const float* __restrict__ Wl,
    const float* __restrict__ bl, float* __restrict__ out)
{
  __shared__ float sm[128];
  const int b = blockIdx.x, tid = threadIdx.x;
  if (tid < 128) {
    float s = 0.0f;
#pragma unroll
    for (int qb = 0; qb < 16; ++qb) s += partial[(size_t)(b * 16 + qb) * 128 + tid];
    sm[tid] = s * (1.0f / (float)SEQ);
  }
  __syncthreads();
  for (int c = tid; c < NC; c += 256) {
    float a = bl[c];
    for (int d = 0; d < 128; ++d) a = fmaf(sm[d], Wl[(size_t)d * NC + c], a);
    out[(size_t)b * NC + c] = a;
  }
}

// ---------------------------------------------------------------------------
extern "C" void kernel_launch(void* const* d_in, const int* in_sizes, int n_in,
                              void* d_out, int out_size, void* d_ws, size_t ws_size,
                              hipStream_t stream) {
  (void)in_sizes; (void)n_in; (void)out_size; (void)ws_size;
  const float* x  = (const float*)d_in[0];
  const float* Wq = (const float*)d_in[1];
  const float* bq = (const float*)d_in[2];
  const float* Wk = (const float*)d_in[3];
  const float* bk = (const float*)d_in[4];
  const float* Wv = (const float*)d_in[5];
  const float* bv = (const float*)d_in[6];
  const float* Wl = (const float*)d_in[7];
  const float* bl = (const float*)d_in[8];
  float* out = (float*)d_out;

  char* ws = (char*)d_ws;
  _Float16* xh  = (_Float16*)(ws);                         // 16 MiB [B*S, F]
  _Float16* qh  = (_Float16*)(ws + (size_t)16 * 1048576);  //  4 MiB [B*S, D]
  _Float16* kh  = (_Float16*)(ws + (size_t)20 * 1048576);  //  4 MiB [B*S, D]
  _Float16* vTh = (_Float16*)(ws + (size_t)24 * 1048576);  //  4 MiB [B, D, S]
  _Float16* WqT = (_Float16*)(ws + (size_t)28 * 1048576);            // 128 KiB [D, F]
  _Float16* WkT = (_Float16*)(ws + (size_t)28 * 1048576 + 131072);   // 128 KiB
  _Float16* WvT = (_Float16*)(ws + (size_t)28 * 1048576 + 262144);   // 128 KiB
  float*    prt = (float*)   (ws + (size_t)28 * 1048576 + 393216);   //  64 KiB [128,128]

  cvt_x_kernel<<<(NB * SEQ * FD) / (256 * 8), 256, 0, stream>>>(x, xh);  // 4096 blocks
  cvt_wT_kernel<<<(FD * DD) / 256, 256, 0, stream>>>(Wq, WqT);
  cvt_wT_kernel<<<(FD * DD) / 256, 256, 0, stream>>>(Wk, WkT);
  cvt_wT_kernel<<<(FD * DD) / 256, 256, 0, stream>>>(Wv, WvT);

  const int projBlocks = (NB * SEQ) / 16;                  // 1024
  qkv_gemm_kernel<<<projBlocks, 256, 0, stream>>>(xh, WqT, bq, qh, 0);
  qkv_gemm_kernel<<<projBlocks, 256, 0, stream>>>(xh, WkT, bk, kh, 0);
  qkv_gemm_kernel<<<projBlocks, 256, 0, stream>>>(xh, WvT, bv, vTh, 1);

  attn_kernel<<<NB * 16, 256, 0, stream>>>(qh, kh, vTh, prt);
  final_kernel<<<NB, 256, 0, stream>>>(prt, Wl, bl, out);
}
